// BatchNeuralMemory_19516331393467
// MI455X (gfx1250) — matrix operand order repaired and checked by reading, not compiled, not run on hardware
//
#include <hip/hip_runtime.h>

// ---------------------------------------------------------------------------
// CDNA5 (gfx1250) WMMA implementation of the batched neural-memory scan.
// GEMM core: v_wmma_f32_16x16x32_bf16, 64x128 block tile, 8 waves each doing
// 32x32 (4 WMMA / K-step), LDS double-buffered, fragment-major LDS layout so
// fragment builds are ds_load_b128. Staging uses base+int-offset indexing so
// loads lower to global_load (not flat), gathered to registers before commit.
// ---------------------------------------------------------------------------

typedef __attribute__((ext_vector_type(16))) __bf16 v16bf;
typedef __attribute__((ext_vector_type(8)))  __bf16 v8bf;
typedef __attribute__((ext_vector_type(8)))  float  v8f;

#define TBM 64
#define TBN 128
#define TBK 32
#define ASZ (TBM * TBK / 2)   // u32 words per A tile (1024)
#define BSZ (TBN * TBK / 2)   // u32 words per B tile (2048)

__device__ __forceinline__ unsigned bfbits(float f) {
    union { float f; unsigned u; } x; x.f = f;
    return x.u + 0x7FFFu + ((x.u >> 16) & 1u);     // round-to-nearest-even
}
// pack bf16(f0) into [15:0], bf16(f1) into [31:16]
__device__ __forceinline__ unsigned pack2bf(float f0, float f1) {
    return (bfbits(f0) >> 16) | (bfbits(f1) & 0xFFFF0000u);
}

__device__ __forceinline__ float sigmoidf_(float x) { return 1.f / (1.f + __expf(-x)); }
__device__ __forceinline__ float siluf_(float x)    { return x / (1.f + __expf(-x)); }

__device__ __forceinline__ float block_reduce_sum(float v, float* sh) {
    int tid = threadIdx.x;
    sh[tid] = v;
    __syncthreads();
    for (int s = 128; s > 0; s >>= 1) {
        if (tid < s) sh[tid] += sh[tid + s];
        __syncthreads();
    }
    float r = sh[0];
    __syncthreads();
    return r;
}

// ---------------------------------------------------------------------------
// Batched GEMM: C[M,N] = A op B, fp32 in/out, bf16 WMMA core.
//   TA=0: A is [M,K]; TA=1: A is [K,M] (compute A^T @ B)
//   TB=0: B is [K,N]; TB=1: B is [N,K] (compute A @ B^T)
// Requirements (all satisfied here): M%64==0, N%128==0, K%32==0, even ld*.
// Block 256 threads = 8 waves; wave w: rows (w&1)*32, cols (w>>1)*32 of the
// 64x128 tile -> 2 A frags x 2 B frags -> 4 v_wmma per K-step.
// LDS (double buffered): As32[m][kp] = bf16 pair (K=2kp,2kp+1) of row m;
//                        Bt32[n][kp] = same for column n (B^T, K-contiguous).
// ---------------------------------------------------------------------------
template<bool TA, bool TB, bool SILU>
__global__ __launch_bounds__(256) void wmma_gemm(
    const float* __restrict__ A, const float* __restrict__ B, float* __restrict__ C,
    int M, int N, int K, int lda, int ldb, int ldc,
    long long strideA, long long strideB, long long strideC)
{
    __shared__ __align__(16) unsigned lds[2 * (ASZ + BSZ)];   // 24 KB

    const int bz = blockIdx.z;
    A += (long long)bz * strideA;
    B += (long long)bz * strideB;
    C += (long long)bz * strideC;

    const int m0 = blockIdx.y * TBM;
    const int n0 = blockIdx.x * TBN;
    const int tid  = threadIdx.x;
    const int wave = tid >> 5;
    const int lane = tid & 31;
    const int half = lane >> 4;          // 0 / 1
    const int l16  = lane & 15;
    const int wr = (wave & 1) * 32;      // wave row offset in tile
    const int wc = (wave >> 1) * 32;     // wave col offset in tile

    // ---- per-thread staging slots: 4 bf16-pairs of A, 8 of B ----
    long long aoff[4]; int aidx[4];
    long long boff[8]; int bidx[8];
    #pragma unroll
    for (int t = 0; t < 4; ++t) {
        int j = tid + t * 256;                       // A pair slot 0..1023
        int m, kp;
        if (TA) { m = j & 63; kp = j >> 6; }         // coalesce along m
        else    { m = j >> 4; kp = j & 15; }         // coalesce along k (float2)
        aidx[t] = m * (TBK / 2) + kp;
        aoff[t] = TA ? (long long)(2 * kp) * lda + (m0 + m)
                     : (long long)(m0 + m) * lda + 2 * kp;
    }
    #pragma unroll
    for (int t = 0; t < 8; ++t) {
        int j = tid + t * 256;                       // B pair slot 0..2047
        int n, kq;
        if (TB) { n = j >> 4;  kq = j & 15; }        // coalesce along k (float2)
        else    { n = j & 127; kq = j >> 7; }        // coalesce along n
        bidx[t] = n * (TBK / 2) + kq;
        boff[t] = TB ? (long long)(n0 + n) * ldb + 2 * kq
                     : (long long)(2 * kq) * ldb + (n0 + n);
    }
    const long long aStep = TA ? (long long)TBK * lda : (long long)TBK;
    const long long bStep = TB ? (long long)TBK       : (long long)TBK * ldb;

    float2 av[4], bv[8];
    // gather one K-tile into registers (loads stay outstanding together)
    auto gather = [&]() {
        #pragma unroll
        for (int t = 0; t < 4; ++t) {
            if (TA) { av[t].x = A[aoff[t]]; av[t].y = A[aoff[t] + lda]; }
            else    { av[t] = *(const float2*)(A + aoff[t]); }
            aoff[t] += aStep;
        }
        #pragma unroll
        for (int t = 0; t < 8; ++t) {
            if (TB) { bv[t] = *(const float2*)(B + boff[t]); }
            else    { bv[t].x = B[boff[t]]; bv[t].y = B[boff[t] + ldb]; }
            boff[t] += bStep;
        }
    };
    // convert + commit registers to LDS buffer `buf`
    auto commit = [&](int buf) {
        unsigned* As = lds + buf * (ASZ + BSZ);
        unsigned* Bt = As + ASZ;
        #pragma unroll
        for (int t = 0; t < 4; ++t) As[aidx[t]] = pack2bf(av[t].x, av[t].y);
        #pragma unroll
        for (int t = 0; t < 8; ++t) Bt[bidx[t]] = pack2bf(bv[t].x, bv[t].y);
    };

    v8f acc00 = {}, acc01 = {}, acc10 = {}, acc11 = {};

    // ---- fragment compute: 8x ds_load_b128 + 4x v_wmma ----
    auto compute = [&](int buf) {
        const unsigned* As = lds + buf * (ASZ + BSZ);
        const unsigned* Bt = As + ASZ;
        // A 16x32 bf16: lane<16 -> elems 0..7 = K 0..7, 8..15 = K 16..23
        //               lane>=16 -> K 8..15 and K 24..31    (row = l16)
        union { v16bf v; v8bf h[2]; } a0, a1, b0, b1;
        const int ar = (wr + l16) * (TBK / 2);
        a0.h[0] = *(const v8bf*)&As[ar + half * 4];
        a0.h[1] = *(const v8bf*)&As[ar + 8 + half * 4];
        a1.h[0] = *(const v8bf*)&As[ar + 16 * (TBK / 2) + half * 4];
        a1.h[1] = *(const v8bf*)&As[ar + 16 * (TBK / 2) + 8 + half * 4];
        // B 32x16 bf16: same K order as A (lane<16: K 0..7,16..23; lane>=16: K 8..15,24..31)
        const int br = (wc + l16) * (TBK / 2);
        b0.h[0] = *(const v8bf*)&Bt[br + half * 4];
        b0.h[1] = *(const v8bf*)&Bt[br + 8 + half * 4];
        b1.h[0] = *(const v8bf*)&Bt[br + 16 * (TBK / 2) + half * 4];
        b1.h[1] = *(const v8bf*)&Bt[br + 16 * (TBK / 2) + 8 + half * 4];

        acc00 = __builtin_amdgcn_wmma_f32_16x16x32_bf16(false, a0.v, false, b0.v,
                                                        (short)0, acc00, false, false);
        acc01 = __builtin_amdgcn_wmma_f32_16x16x32_bf16(false, a0.v, false, b1.v,
                                                        (short)0, acc01, false, false);
        acc10 = __builtin_amdgcn_wmma_f32_16x16x32_bf16(false, a1.v, false, b0.v,
                                                        (short)0, acc10, false, false);
        acc11 = __builtin_amdgcn_wmma_f32_16x16x32_bf16(false, a1.v, false, b1.v,
                                                        (short)0, acc11, false, false);
    };

    // ---- double-buffered main loop: one barrier per K-step ----
    const int T = K / TBK;
    gather();
    commit(0);
    __syncthreads();
    for (int t = 0; t < T; ++t) {
        int cur = t & 1;
        if (t + 1 < T) {
            gather();                                   // overlaps with WMMA below
            __builtin_prefetch(A + aoff[0], 0, 1);      // gfx1250 global_prefetch_b8
            __builtin_prefetch(B + boff[0], 0, 1);
        }
        compute(cur);
        if (t + 1 < T) {
            commit(cur ^ 1);
            __syncthreads();
        }
    }

    // D layout: VGPR j -> row = 8*half + j, col = l16
    const int row0 = m0 + wr + half * 8;
    const int col0 = n0 + wc + l16;
    #pragma unroll
    for (int j = 0; j < 8; ++j) {
        float v00 = acc00[j], v01 = acc01[j], v10 = acc10[j], v11 = acc11[j];
        if (SILU) {
            v00 = siluf_(v00); v01 = siluf_(v01);
            v10 = siluf_(v10); v11 = siluf_(v11);
        }
        C[(long long)(row0 + j) * ldc + col0]           = v00;
        C[(long long)(row0 + j) * ldc + col0 + 16]      = v01;
        C[(long long)(row0 + 16 + j) * ldc + col0]      = v10;
        C[(long long)(row0 + 16 + j) * ldc + col0 + 16] = v11;
    }
}

// ---------------------------------------------------------------------------
// Gating: per (b,chunk) dot(xc, {alpha,theta,eta}_w) -> beta/theta/eta scalars
// ---------------------------------------------------------------------------
__global__ __launch_bounds__(256) void gate_kernel(
    const float* __restrict__ x, const float* __restrict__ aw,
    const float* __restrict__ tw, const float* __restrict__ ew,
    float* __restrict__ theta, float* __restrict__ eta, float* __restrict__ beta,
    int CH)
{
    const float* xc = x + (long long)blockIdx.x * CH;
    float sa = 0.f, st = 0.f, se = 0.f;
    for (int i = threadIdx.x; i < CH; i += 256) {
        float xv = xc[i];
        sa += xv * aw[i]; st += xv * tw[i]; se += xv * ew[i];
    }
    __shared__ float ra[256], rt[256], re[256];
    ra[threadIdx.x] = sa; rt[threadIdx.x] = st; re[threadIdx.x] = se;
    __syncthreads();
    for (int s = 128; s > 0; s >>= 1) {
        if (threadIdx.x < s) {
            ra[threadIdx.x] += ra[threadIdx.x + s];
            rt[threadIdx.x] += rt[threadIdx.x + s];
            re[threadIdx.x] += re[threadIdx.x + s];
        }
        __syncthreads();
    }
    if (threadIdx.x == 0) {
        theta[blockIdx.x] = sigmoidf_(rt[0]) * 0.01f;   // BASE_LR
        eta[blockIdx.x]   = sigmoidf_(re[0]);
        beta[blockIdx.x]  = 1.f - sigmoidf_(ra[0]);
    }
}

__global__ __launch_bounds__(256) void silu_kernel(
    const float* __restrict__ in, float* __restrict__ out, int n)
{
    int i = blockIdx.x * 256 + threadIdx.x;
    if (i < n) out[i] = siluf_(in[i]);
}

// rmsnorm over rows with shared weight [H]
__global__ __launch_bounds__(256) void rmsnorm_kernel(
    const float* __restrict__ in, const float* __restrict__ w,
    float* __restrict__ out, int H)
{
    __shared__ float sh[256];
    const float* xr = in + (long long)blockIdx.x * H;
    float ss = 0.f;
    for (int i = threadIdx.x; i < H; i += 256) { float t = xr[i]; ss += t * t; }
    ss = block_reduce_sum(ss, sh);
    float rs = rsqrtf(ss / H + 1e-6f);
    float* orow = out + (long long)blockIdx.x * H;
    for (int i = threadIdx.x; i < H; i += 256) orow[i] = xr[i] * rs * w[i];
}

// per row: rs = rsqrt(mean(y^2)+eps); dpred = (2*theta/H)*(kc + y*rs*ln - vc)
__global__ __launch_bounds__(256) void pred_err_kernel(
    const float* __restrict__ y, const float* __restrict__ kbuf,
    const float* __restrict__ vbuf, const float* __restrict__ ln,
    const float* __restrict__ theta, int step, int NC, int S, int Crows, int H,
    float* __restrict__ dpred, float* __restrict__ rs_out)
{
    __shared__ float sh[256];
    int row = blockIdx.x;
    int b = row / Crows, c = row % Crows;
    const float* yr = y + (long long)row * H;
    long long koff = ((long long)b * S + (long long)step * Crows + c) * H;
    float ss = 0.f;
    for (int i = threadIdx.x; i < H; i += 256) { float t = yr[i]; ss += t * t; }
    ss = block_reduce_sum(ss, sh);
    float rs = rsqrtf(ss / H + 1e-6f);
    if (threadIdx.x == 0) rs_out[row] = rs;
    float sc = 2.f * theta[b * NC + step] / H;
    float* drow = dpred + (long long)row * H;
    const float* lr = ln + (long long)b * H;
    for (int i = threadIdx.x; i < H; i += 256) {
        float pred = kbuf[koff + i] + yr[i] * rs * lr[i];
        drow[i] = sc * (pred - vbuf[koff + i]);
    }
}

// rmsnorm backward wrt y: t=g*w; dy = rs*t - y*rs^3*sum(t*y)/H
__global__ __launch_bounds__(256) void norm_bwd_kernel(
    const float* __restrict__ dpred, const float* __restrict__ y,
    const float* __restrict__ rs, const float* __restrict__ ln,
    float* __restrict__ dy, int H, int Crows)
{
    __shared__ float sh[256];
    int row = blockIdx.x;
    int b = row / Crows;
    const float* gr = dpred + (long long)row * H;
    const float* yr = y + (long long)row * H;
    const float* lr = ln + (long long)b * H;
    float r = rs[row];
    float s = 0.f;
    for (int i = threadIdx.x; i < H; i += 256) s += gr[i] * lr[i] * yr[i];
    s = block_reduce_sum(s, sh);
    float coef = r * r * r * s / H;
    float* drow = dy + (long long)row * H;
    for (int i = threadIdx.x; i < H; i += 256)
        drow[i] = r * gr[i] * lr[i] - yr[i] * coef;
}

// dln[b,h] = sum_c dpred[b,c,h]*y[b,c,h]*rs[b,c]
__global__ __launch_bounds__(256) void dln_kernel(
    const float* __restrict__ dpred, const float* __restrict__ y,
    const float* __restrict__ rs, float* __restrict__ dln, int H, int Crows)
{
    int idx = blockIdx.x * 256 + threadIdx.x;    // b*H + h
    int b = idx / H, h = idx % H;
    float s = 0.f;
    for (int c = 0; c < Crows; ++c) {
        long long r = (long long)b * Crows + c;
        s += dpred[r * H + h] * y[r * H + h] * rs[r];
    }
    dln[idx] = s;
}

// dpre0 = dh * silu'(z0)
__global__ __launch_bounds__(256) void silu_bwd_kernel(
    const float* __restrict__ dh, const float* __restrict__ z,
    float* __restrict__ out, int n)
{
    int i = blockIdx.x * 256 + threadIdx.x;
    if (i < n) {
        float zi = z[i];
        float sg = sigmoidf_(zi);
        out[i] = dh[i] * sg * (1.f + zi * (1.f - sg));
    }
}

// per-batch sum of squares over {dW0, dW1, dln}
__global__ __launch_bounds__(256) void sqsum_kernel(
    const float* __restrict__ dW0, const float* __restrict__ dW1,
    const float* __restrict__ dln, float* __restrict__ sqacc,
    long long perW, int Hn)
{
    __shared__ float sh[256];
    int b = blockIdx.y;
    long long total = 2 * perW + Hn;
    float s = 0.f;
    for (long long i = (long long)blockIdx.x * 256 + threadIdx.x; i < total;
         i += (long long)gridDim.x * 256) {
        float g;
        if (i < perW)               g = dW0[(long long)b * perW + i];
        else if (i < 2 * perW)      g = dW1[(long long)b * perW + (i - perW)];
        else                        g = dln[(long long)b * Hn + (i - 2 * perW)];
        s += g * g;
    }
    s = block_reduce_sum(s, sh);
    if (threadIdx.x == 0) atomicAdd(&sqacc[b], s);
}

// fused clip + momentum + weight decay:  g*=clip; sur=eta*sur-g; mem=beta*mem+sur
__global__ __launch_bounds__(256) void update_kernel(
    float* __restrict__ mem, float* __restrict__ sur, const float* __restrict__ grad,
    const float* __restrict__ sqacc, const float* __restrict__ eta,
    const float* __restrict__ beta, int step, int NC, long long perBatch)
{
    int b = blockIdx.y;
    float clip = fminf(1.f / (sqrtf(sqacc[b]) + 1e-6f), 1.f);
    float et = eta[b * NC + step];
    float be = beta[b * NC + step];
    for (long long i = (long long)blockIdx.x * 256 + threadIdx.x; i < perBatch;
         i += (long long)gridDim.x * 256) {
        long long o = (long long)b * perBatch + i;
        float g = grad[o] * clip;
        float s = et * sur[o] - g;
        sur[o] = s;
        mem[o] = be * mem[o] + s;
    }
}

// out = x + rmsnorm(y, ln[b])  (per-batch ln)
__global__ __launch_bounds__(256) void addnorm_kernel(
    const float* __restrict__ y, const float* __restrict__ x,
    const float* __restrict__ ln, float* __restrict__ out, int H, int rowsPerBatch)
{
    __shared__ float sh[256];
    int row = blockIdx.x;
    int b = row / rowsPerBatch;
    const float* yr = y + (long long)row * H;
    float ss = 0.f;
    for (int i = threadIdx.x; i < H; i += 256) { float t = yr[i]; ss += t * t; }
    ss = block_reduce_sum(ss, sh);
    float rs = rsqrtf(ss / H + 1e-6f);
    const float* lr = ln + (long long)b * H;
    const float* xr = x + (long long)row * H;
    float* orow = out + (long long)row * H;
    for (int i = threadIdx.x; i < H; i += 256)
        orow[i] = xr[i] + yr[i] * rs * lr[i];
}

// ---------------------------------------------------------------------------
extern "C" void kernel_launch(void* const* d_in, const int* in_sizes, int n_in,
                              void* d_out, int out_size, void* d_ws, size_t ws_size,
                              hipStream_t stream)
{
    (void)in_sizes; (void)n_in; (void)out_size; (void)ws_size;
    const int B = 4, S = 1024, H = 1024, I = 1024, C = 64, NC = S / C;

    const float* x   = (const float*)d_in[0];
    const float* wq  = (const float*)d_in[1];
    const float* wk  = (const float*)d_in[2];
    const float* wv  = (const float*)d_in[3];
    const float* qnw = (const float*)d_in[4];
    const float* knw = (const float*)d_in[5];
    const float* aw  = (const float*)d_in[6];
    const float* tw  = (const float*)d_in[7];
    const float* ew  = (const float*)d_in[8];
    const float* mW0 = (const float*)d_in[9];
    const float* mW1 = (const float*)d_in[10];
    const float* mln = (const float*)d_in[11];
    float* out = (float*)d_out;

    float* p = (float*)d_ws;
    auto alloc = [&](size_t n) { float* r = p; p += n; return r; };
    const size_t NSH = (size_t)B * S * H;   // 4M floats
    const size_t NW  = (size_t)B * I * H;   // 4M floats

    float* kbuf  = alloc(NSH);              // reused as q after the scan
    float* vbuf  = alloc(NSH);
    float* W0    = alloc(NW);
    float* W1    = alloc(NW);
    float* lnw   = alloc((size_t)B * H);
    float* sW0   = alloc(NW);
    float* sW1   = alloc(NW);
    float* sln   = alloc((size_t)B * H);
    float* dW0   = alloc(NW);               // reused as h_ret
    float* dW1   = alloc(NW);               // reused as y_ret
    float* dln   = alloc((size_t)B * H);
    float* z0    = alloc((size_t)B * C * I);
    float* hbuf  = alloc((size_t)B * C * I);
    float* ybuf  = alloc((size_t)B * C * H);
    float* dpred = alloc((size_t)B * C * H);
    float* dybuf = alloc((size_t)B * C * H);
    float* dhbuf = alloc((size_t)B * C * I);
    float* dpre0 = alloc((size_t)B * C * I);
    float* rsbuf = alloc((size_t)B * C);
    float* theta = alloc((size_t)B * NC);
    float* eta   = alloc((size_t)B * NC);
    float* beta  = alloc((size_t)B * NC);
    float* sqacc = alloc((size_t)B);

    // working copies of fast weights (inputs must not be mutated); zero surprise
    hipMemcpyAsync(W0, mW0, NW * sizeof(float), hipMemcpyDeviceToDevice, stream);
    hipMemcpyAsync(W1, mW1, NW * sizeof(float), hipMemcpyDeviceToDevice, stream);
    hipMemcpyAsync(lnw, mln, (size_t)B * H * sizeof(float), hipMemcpyDeviceToDevice, stream);
    hipMemsetAsync(sW0, 0, NW * sizeof(float), stream);
    hipMemsetAsync(sW1, 0, NW * sizeof(float), stream);
    hipMemsetAsync(sln, 0, (size_t)B * H * sizeof(float), stream);

    dim3 blk(256);

    // gating scalars
    gate_kernel<<<B * NC, blk, 0, stream>>>(x, aw, tw, ew, theta, eta, beta, C * H);

    // k = rmsnorm(silu(x@wk^T)); v = silu(x@wv^T)
    dim3 gproj(H / TBN, (B * S) / TBM, 1);
    wmma_gemm<false, true, true><<<gproj, blk, 0, stream>>>(
        x, wk, kbuf, B * S, H, H, H, H, H, 0, 0, 0);
    rmsnorm_kernel<<<B * S, blk, 0, stream>>>(kbuf, knw, kbuf, H);
    wmma_gemm<false, true, true><<<gproj, blk, 0, stream>>>(
        x, wv, vbuf, B * S, H, H, H, H, H, 0, 0, 0);

    const long long sSH = (long long)S * H, sIH = (long long)I * H;
    const long long sCI = (long long)C * I, sCH = (long long)C * H;

    for (int step = 0; step < NC; ++step) {
        const float* kc = kbuf + (size_t)step * C * H;   // batch stride S*H
        // z0 = kc @ W0^T   [C x I]
        wmma_gemm<false, true, false><<<dim3(I / TBN, 1, B), blk, 0, stream>>>(
            kc, W0, z0, C, I, H, H, H, I, sSH, sIH, sCI);
        silu_kernel<<<(B * C * I + 255) / 256, blk, 0, stream>>>(z0, hbuf, B * C * I);
        // y = h @ W1^T     [C x H]
        wmma_gemm<false, true, false><<<dim3(H / TBN, 1, B), blk, 0, stream>>>(
            hbuf, W1, ybuf, C, H, I, I, I, H, sCI, sIH, sCH);
        pred_err_kernel<<<B * C, blk, 0, stream>>>(
            ybuf, kbuf, vbuf, lnw, theta, step, NC, S, C, H, dpred, rsbuf);
        norm_bwd_kernel<<<B * C, blk, 0, stream>>>(dpred, ybuf, rsbuf, lnw, dybuf, H, C);
        dln_kernel<<<(B * H) / 256, blk, 0, stream>>>(dpred, ybuf, rsbuf, dln, H, C);
        // dh = dy @ W1     [C x I]
        wmma_gemm<false, false, false><<<dim3(I / TBN, 1, B), blk, 0, stream>>>(
            dybuf, W1, dhbuf, C, I, H, H, I, I, sCH, sIH, sCI);
        silu_bwd_kernel<<<(B * C * I + 255) / 256, blk, 0, stream>>>(dhbuf, z0, dpre0, B * C * I);
        // dW1 = dy^T @ h   [H x I]
        wmma_gemm<true, false, false><<<dim3(I / TBN, H / TBM, B), blk, 0, stream>>>(
            dybuf, hbuf, dW1, H, I, C, H, I, I, sCH, sCI, sIH);
        // dW0 = dpre0^T @ kc [I x H]
        wmma_gemm<true, false, false><<<dim3(H / TBN, I / TBM, B), blk, 0, stream>>>(
            dpre0, kc, dW0, I, H, C, I, H, H, sCI, sSH, sIH);
        // per-sample global grad-norm clip + momentum + decay update
        hipMemsetAsync(sqacc, 0, B * sizeof(float), stream);
        sqsum_kernel<<<dim3(128, B), blk, 0, stream>>>(dW0, dW1, dln, sqacc,
                                                       (long long)H * I, H);
        update_kernel<<<dim3(256, B), blk, 0, stream>>>(W0, sW0, dW0, sqacc, eta, beta,
                                                        step, NC, (long long)H * I);
        update_kernel<<<dim3(256, B), blk, 0, stream>>>(W1, sW1, dW1, sqacc, eta, beta,
                                                        step, NC, (long long)H * I);
        update_kernel<<<dim3(4, B), blk, 0, stream>>>(lnw, sln, dln, sqacc, eta, beta,
                                                      step, NC, (long long)H);
    }

    // retrieval with final fast weights
    float* qbuf = kbuf;
    wmma_gemm<false, true, true><<<gproj, blk, 0, stream>>>(
        x, wq, qbuf, B * S, H, H, H, H, H, 0, 0, 0);
    rmsnorm_kernel<<<B * S, blk, 0, stream>>>(qbuf, qnw, qbuf, H);
    float* hret = dW0;
    float* yret = dW1;
    wmma_gemm<false, true, true><<<dim3(I / TBN, S / TBM, B), blk, 0, stream>>>(
        qbuf, W0, hret, S, I, H, H, H, I, sSH, sIH, (long long)S * I);
    wmma_gemm<false, true, false><<<dim3(H / TBN, S / TBM, B), blk, 0, stream>>>(
        hret, W1, yret, S, H, I, I, I, H, (long long)S * I, sIH, sSH);
    addnorm_kernel<<<B * S, blk, 0, stream>>>(yret, qbuf, lnw, out, H, S);
}
